// GCN_48533130445252
// MI455X (gfx1250) — matrix operand order repaired and checked by reading, not compiled, not run on hardware
//
#include <hip/hip_runtime.h>

typedef __attribute__((ext_vector_type(16))) _Float16 v16h;
typedef __attribute__((ext_vector_type(8)))  _Float16 v8h;
typedef __attribute__((ext_vector_type(4)))  _Float16 v4h;
typedef __attribute__((ext_vector_type(8)))  float    v8f;

#define C_DIM 128
#define LDSTR 136   // row stride in halves: 272B = 17*16B (16B-aligned rows)

// ---------------- degree / normalization ----------------
__global__ void k_init_deg(float* __restrict__ deg, int n) {
    int i = blockIdx.x * blockDim.x + threadIdx.x;
    if (i < n) deg[i] = 1.0f;  // self-loop
}

__global__ void k_deg_edges(const int* __restrict__ dst, float* __restrict__ deg, int ne) {
    int e = blockIdx.x * blockDim.x + threadIdx.x;
    if (e < ne)
        __hip_atomic_fetch_add(&deg[dst[e]], 1.0f, __ATOMIC_RELAXED,
                               __HIP_MEMORY_SCOPE_AGENT);
}

__global__ void k_rsqrt(float* __restrict__ degv, int n) {
    int i = blockIdx.x * blockDim.x + threadIdx.x;
    if (i < n) degv[i] = rsqrtf(degv[i]);  // deg >= 1
}

// per-edge symmetric norm, shared by both layers
__global__ void k_edge_norm(const int* __restrict__ src, const int* __restrict__ dst,
                            const float* __restrict__ dinv, float* __restrict__ nrm, int ne) {
    int e = blockIdx.x * blockDim.x + threadIdx.x;
    if (e < ne) nrm[e] = dinv[src[e]] * dinv[dst[e]];
}

// ---- GEMM + fused aggregation-init epilogue ----
// H   = A(nrows x 128) @ W(128 x 128)     (optional ReLU on A)
// OUT = H * dinv[row]^2 + bias[col]       (self-loop + bias init for the scatter phase)
__global__ void __launch_bounds__(256)
k_gemm_wmma(const float* __restrict__ A, const float* __restrict__ W,
            float* __restrict__ H, float* __restrict__ OUT,
            const float* __restrict__ dinv, const float* __restrict__ bias,
            int nrows, int relu_in) {
    __shared__ __attribute__((aligned(16))) _Float16 sA[128][LDSTR];  // rows x K
    __shared__ __attribute__((aligned(16))) _Float16 sB[128][LDSTR];  // cols x K (W transposed)

    const int row0 = blockIdx.x * 128;

    // stage A strip: float4 loads, packed v4h LDS stores (fused ReLU for layer 2)
    for (int idx = threadIdx.x; idx < 128 * 32; idx += 256) {
        int r = idx >> 5, c4 = (idx & 31) << 2;
        int gr = row0 + r;
        float4 v = make_float4(0.f, 0.f, 0.f, 0.f);
        if (gr < nrows) v = *(const float4*)(A + (size_t)gr * C_DIM + c4);
        if (relu_in) {
            v.x = fmaxf(v.x, 0.f); v.y = fmaxf(v.y, 0.f);
            v.z = fmaxf(v.z, 0.f); v.w = fmaxf(v.w, 0.f);
        }
        v4h p = { (_Float16)v.x, (_Float16)v.y, (_Float16)v.z, (_Float16)v.w };
        *(v4h*)&sA[r][c4] = p;
    }
    // stage W transposed: sB[n][k] = W[k][n] -> contiguous B-fragment reads
    for (int idx = threadIdx.x; idx < 128 * 32; idx += 256) {
        int k = idx >> 5, c4 = (idx & 31) << 2;
        float4 v = *(const float4*)(W + (size_t)k * C_DIM + c4);
        sB[c4 + 0][k] = (_Float16)v.x;
        sB[c4 + 1][k] = (_Float16)v.y;
        sB[c4 + 2][k] = (_Float16)v.z;
        sB[c4 + 3][k] = (_Float16)v.w;
    }
    __syncthreads();

    const int lane = threadIdx.x & 31;
    const int wid  = threadIdx.x >> 5;   // 8 waves: wave -> 16-row strip
    const int r    = lane & 15;
    const int half = lane >> 4;
    const int arow = wid * 16 + r;

    // preload the 4 A fragments for this wave's row strip (reused by all 8 col tiles)
    v16h af[4];
    #pragma unroll
    for (int kt = 0; kt < 4; ++kt) {
        const int kb = kt * 32;
        v8h a0 = *(const v8h*)&sA[arow][kb + half * 8];
        v8h a1 = *(const v8h*)&sA[arow][kb + 16 + half * 8];
        af[kt] = __builtin_shufflevector(a0, a1, 0, 1, 2, 3, 4, 5, 6, 7,
                                                 8, 9, 10, 11, 12, 13, 14, 15);
    }

    // 32 back-to-back WMMAs over 8 column tiles x 4 K-chunks
    v8f acc[8];
    #pragma unroll
    for (int ct = 0; ct < 8; ++ct) acc[ct] = (v8f){};
    #pragma unroll
    for (int ct = 0; ct < 8; ++ct) {
        const int bcol = ct * 16 + r;
        #pragma unroll
        for (int kt = 0; kt < 4; ++kt) {
            const int kb = kt * 32 + half * 8;    // lane's K: [kb, kb+8) and [kb+16, kb+24)
            v8h b0 = *(const v8h*)&sB[bcol][kb];
            v8h b1 = *(const v8h*)&sB[bcol][kb + 16];
            v16h b = __builtin_shufflevector(b0, b1, 0, 1, 2, 3, 4, 5, 6, 7,
                                                     8, 9, 10, 11, 12, 13, 14, 15);
            acc[ct] = __builtin_amdgcn_wmma_f32_16x16x32_f16(
                false, af[kt], false, b, (short)0, acc[ct], false, false);
        }
    }

    // epilogue: C/D layout -> lane N = lane&15, VGPR v -> M = v + 8*half
    const int rbase = row0 + wid * 16 + half * 8;
    if (row0 + 128 <= nrows) {               // common path: whole block in range
        float di2[8];
        #pragma unroll
        for (int v = 0; v < 8; ++v) { float di = dinv[rbase + v]; di2[v] = di * di; }
        #pragma unroll
        for (int ct = 0; ct < 8; ++ct) {
            const int bcol = ct * 16 + r;
            const float bc = bias[bcol];
            #pragma unroll
            for (int v = 0; v < 8; ++v) {
                size_t o = (size_t)(rbase + v) * C_DIM + bcol;
                float hv = acc[ct][v];
                H[o]   = hv;
                OUT[o] = hv * di2[v] + bc;
            }
        }
    } else {                                  // only the last ragged block
        #pragma unroll
        for (int ct = 0; ct < 8; ++ct) {
            const int bcol = ct * 16 + r;
            const float bc = bias[bcol];
            #pragma unroll
            for (int v = 0; v < 8; ++v) {
                int gr = rbase + v;
                if (gr < nrows) {
                    float di = dinv[gr];
                    size_t o = (size_t)gr * C_DIM + bcol;
                    float hv = acc[ct][v];
                    H[o]   = hv;
                    OUT[o] = hv * di * di + bc;
                }
            }
        }
    }
}

// ---------------- edge scatter: one wave per edge ----------------
// coalesced 512B row gather + 512B of L2-resident fp32 atomics
__global__ void __launch_bounds__(256)
k_agg_edges(const float* __restrict__ h, const float* __restrict__ nrmv,
            const int* __restrict__ src, const int* __restrict__ dst,
            float* __restrict__ out, int ne) {
    int gid = blockIdx.x * blockDim.x + threadIdx.x;
    int e = gid >> 5;
    if (e >= ne) return;
    int c = (gid & 31) << 2;             // 4 channels per lane
    int s = src[e], d = dst[e];
    float nrm = nrmv[e];
    const float4 hv = *reinterpret_cast<const float4*>(h + (size_t)s * C_DIM + c);
    float* o = out + (size_t)d * C_DIM + c;
    __hip_atomic_fetch_add(o + 0, hv.x * nrm, __ATOMIC_RELAXED, __HIP_MEMORY_SCOPE_AGENT);
    __hip_atomic_fetch_add(o + 1, hv.y * nrm, __ATOMIC_RELAXED, __HIP_MEMORY_SCOPE_AGENT);
    __hip_atomic_fetch_add(o + 2, hv.z * nrm, __ATOMIC_RELAXED, __HIP_MEMORY_SCOPE_AGENT);
    __hip_atomic_fetch_add(o + 3, hv.w * nrm, __ATOMIC_RELAXED, __HIP_MEMORY_SCOPE_AGENT);
}

// ---------------- driver ----------------
extern "C" void kernel_launch(void* const* d_in, const int* in_sizes, int n_in,
                              void* d_out, int out_size, void* d_ws, size_t ws_size,
                              hipStream_t stream) {
    const float* x  = (const float*)d_in[0];
    const int*   ei = (const int*)d_in[1];   // [2, E]: row0 = src, row1 = dst
    const float* W1 = (const float*)d_in[2];
    const float* b1 = (const float*)d_in[3];
    const float* W2 = (const float*)d_in[4];
    const float* b2 = (const float*)d_in[5];
    float* out = (float*)d_out;

    const int n  = in_sizes[0] / C_DIM;   // 50000
    const int ne = in_sizes[1] / 2;       // 800000
    const int* src = ei;
    const int* dst = ei + ne;

    char* ws = (char*)d_ws;
    float* degv = (float*)ws;                                   // n floats (deg -> dinv in place)
    float* nrmv = (float*)(ws + (size_t)(1u << 20));            // ne floats
    float* h    = (float*)(ws + (size_t)(8u << 20));            // n*128 floats
    float* agg  = (float*)(ws + (size_t)(8u << 20) + (size_t)(32u << 20)); // n*128 floats

    dim3 blk(256);
    int gN     = (n + 255) / 256;
    int gE     = (ne + 255) / 256;
    int gE32   = (int)(((size_t)ne * 32 + 255) / 256);
    int gTiles = (n + 127) / 128;

    // degrees + symmetric normalization (shared by both layers)
    k_init_deg <<<gN, blk, 0, stream>>>(degv, n);
    k_deg_edges<<<gE, blk, 0, stream>>>(dst, degv, ne);
    k_rsqrt    <<<gN, blk, 0, stream>>>(degv, n);
    k_edge_norm<<<gE, blk, 0, stream>>>(src, dst, degv, nrmv, ne);

    // layer 1: h = x @ W1 ; agg initialized with self-loop + b1 in the epilogue
    k_gemm_wmma<<<gTiles, blk, 0, stream>>>(x, W1, h, agg, degv, b1, n, /*relu_in=*/0);
    k_agg_edges<<<gE32,   blk, 0, stream>>>(h, nrmv, src, dst, agg, ne);

    // layer 2: h2 = relu(agg) @ W2 (ReLU fused into A staging); out init in epilogue
    k_gemm_wmma<<<gTiles, blk, 0, stream>>>(agg, W2, h, out, degv, b2, n, /*relu_in=*/1);
    k_agg_edges<<<gE32,   blk, 0, stream>>>(h, nrmv, src, dst, out, ne);
}
